// Composition_Attention_53541062312244
// MI455X (gfx1250) — compile-verified, matrix operand order checked by reading, not run on hardware
//
#include <hip/hip_runtime.h>

typedef __attribute__((ext_vector_type(16))) _Float16     v16h;
typedef __attribute__((ext_vector_type(8)))  float        v8f;
typedef __attribute__((ext_vector_type(4)))  unsigned int v4u;

#define NEURONS   64
#define GLOBAL_D  103
#define HID       32
#define KPAD      192   // 64 (x) + 128 (padded global feat) = 6 * 32
#define GFPAD     128
#define KSTRIDE   200   // LDS row stride in halfs (bank-conflict-free b128 reads)
#define W1S_ELEMS (6 * 2 * 32 * 16)  // ksteps * ntiles * lanes * halfs

union Frag16 { v16h v; v4u q[2]; };
union H4 { _Float16 h[4]; unsigned long long u64; };

// stable softplus with hardware trans ops: v_exp_f32 / v_log_f32
__device__ inline float softplusf(float z) {
  float t = __expf(-fabsf(z));        // in (0, 1]
  return fmaxf(z, 0.0f) + __logf(1.0f + t);
}

__device__ inline void atomicMaxFloatBits(unsigned int* addr, float v) {
  unsigned int b = __float_as_uint(v);
  if (v >= 0.0f) atomicMax((int*)addr, (int)b);
  else           atomicMin(addr, b);
}

// ---------------------------------------------------------------------------
// Prep: init segmax/segsum, repack W1 into WMMA B-fragment layout (f16),
// convert global_fea -> zero-padded f16 [B,128]
// ---------------------------------------------------------------------------
__global__ void ca_prep_kernel(const float* __restrict__ W1,
                               const float* __restrict__ gfea,
                               _Float16* __restrict__ w1s,
                               _Float16* __restrict__ gfeat,
                               unsigned int* __restrict__ segmax_bits,
                               float* __restrict__ segsum,
                               int Bseg)
{
  int tid = blockIdx.x * blockDim.x + threadIdx.x;
  if (tid < Bseg) {
    segmax_bits[tid] = 0xFF800000u;   // -inf
    segsum[tid] = 0.0f;
    return;
  }
  tid -= Bseg;
  if (tid < W1S_ELEMS) {
    // layout: [t(6)][n(2)][lane(32)][e(16)] halfs, contiguous
    int e    = tid & 15;
    int lane = (tid >> 4) & 31;
    int tn   = tid >> 9;           // t*2 + n
    int n    = tn & 1;
    int t    = tn >> 1;
    int hi   = lane >> 4;
    // 16-bit matrix fragment layout: elems 0..7 -> K = 8*hi + e,
    // elems 8..15 -> K = 16 + 8*hi + (e-8)   (within the 32-wide K-step)
    int k = 32 * t + ((e < 8) ? (8 * hi + e) : (16 + 8 * hi + (e - 8)));
    int c = 16 * n + (lane & 15);
    float v = (k < NEURONS + GLOBAL_D) ? W1[k * HID + c] : 0.0f;
    w1s[tid] = (_Float16)v;
    return;
  }
  tid -= W1S_ELEMS;
  if (tid < Bseg * GFPAD) {
    int row = tid >> 7;
    int col = tid & 127;
    float v = (col < GLOBAL_D) ? gfea[row * GLOBAL_D + col] : 0.0f;
    gfeat[tid] = (_Float16)v;
  }
}

// ---------------------------------------------------------------------------
// Main: per-wave 16-node tile. LDS-staged A (f16), WMMA f32_16x16x32_f16
// h = softplus(chunk@W1 + b1); logit = h@W2 + b2; seg-max via float atomics.
// ---------------------------------------------------------------------------
__global__ __launch_bounds__(256) void ca_logits_kernel(
    const float* __restrict__ x,
    const long long* __restrict__ nb,
    const _Float16* __restrict__ gfeat,
    const _Float16* __restrict__ w1s,
    const float* __restrict__ b1,
    const float* __restrict__ W2,
    const float* __restrict__ b2,
    float* __restrict__ logits,
    unsigned int* __restrict__ segmax_bits,
    int N)
{
  __shared__ _Float16 lds[8][16][KSTRIDE];

  const int wave = threadIdx.x >> 5;
  const int lane = threadIdx.x & 31;
  const int hi   = lane >> 4;
  const int l15  = lane & 15;

  const long long base = ((long long)blockIdx.x * 8 + wave) * 16;
  const long long Nm1  = (long long)N - 1;

  // ---- stage x rows (f32 -> f16) into LDS cols [0,64) ----
  #pragma unroll
  for (int it = 0; it < 8; ++it) {
    int j   = it * 32 + lane;      // float4 chunk id, 0..255
    int r   = j >> 4;              // 0..15
    int col = (j & 15) * 4;        // 0..60
    long long gr = base + r; if (gr > Nm1) gr = Nm1;
    float4 f = *(const float4*)(x + gr * (long long)NEURONS + col);
    H4 h;
    h.h[0] = (_Float16)f.x; h.h[1] = (_Float16)f.y;
    h.h[2] = (_Float16)f.z; h.h[3] = (_Float16)f.w;
    *(unsigned long long*)&lds[wave][r][col] = h.u64;
  }
  // ---- gather per-node graph features (f16, prepadded) into cols [64,192) ----
  #pragma unroll
  for (int it = 0; it < 8; ++it) {
    int r = it * 2 + hi;           // 0..15
    long long gr = base + r; if (gr > Nm1) gr = Nm1;
    long long g = nb[gr];
    const v4u* src = (const v4u*)(gfeat + g * GFPAD);
    *(v4u*)&lds[wave][r][64 + l15 * 8] = src[l15];
  }
  __syncthreads();

  // ---- 16x192 @ 192x32 via 6 K-steps x 2 N-tiles of WMMA ----
  v8f acc0 = {};
  v8f acc1 = {};
  const v4u* bq = (const v4u*)w1s;
  #pragma unroll
  for (int t = 0; t < 6; ++t) {
    Frag16 a, bA, bB;
    const int ka = 32 * t + 8 * hi;
    a.q[0] = *(const v4u*)&lds[wave][l15][ka];
    a.q[1] = *(const v4u*)&lds[wave][l15][ka + 16];
    const int bi0 = ((t * 2 + 0) * 32 + lane) * 2;
    bA.q[0] = bq[bi0];     bA.q[1] = bq[bi0 + 1];
    const int bi1 = ((t * 2 + 1) * 32 + lane) * 2;
    bB.q[0] = bq[bi1];     bB.q[1] = bq[bi1 + 1];
    acc0 = __builtin_amdgcn_wmma_f32_16x16x32_f16(false, a.v, false, bA.v,
                                                  (short)0, acc0, false, false);
    acc1 = __builtin_amdgcn_wmma_f32_16x16x32_f16(false, a.v, false, bB.v,
                                                  (short)0, acc1, false, false);
  }

  // ---- bias + softplus + dot with W2, reduce over 16 lanes ----
  // C/D layout: lane (hi,l15), vgpr v -> row M = 8*hi + v, col N = l15 (+16 for acc1)
  const float b1c0 = b1[l15];
  const float b1c1 = b1[l15 + 16];
  const float w2c0 = W2[l15];
  const float w2c1 = W2[l15 + 16];
  const float b2s  = b2[0];

  float p[8];
  #pragma unroll
  for (int v = 0; v < 8; ++v) {
    float h0 = softplusf(acc0[v] + b1c0);
    float h1 = softplusf(acc1[v] + b1c1);
    p[v] = h0 * w2c0 + h1 * w2c1;
  }
  #pragma unroll
  for (int v = 0; v < 8; ++v) {
    #pragma unroll
    for (int m = 8; m >= 1; m >>= 1) p[v] += __shfl_xor(p[v], m, 32);
    p[v] += b2s;
  }

  // lane0 -> rows base..base+7, lane16 -> rows base+8..base+15
  if (l15 == 0) {
    const long long rbase = base + 8 * hi;
    long long curSeg = -1;
    float curMax = 0.0f;
    #pragma unroll
    for (int v = 0; v < 8; ++v) {
      long long row = rbase + v;
      if (row < N) {
        logits[row] = p[v];
        long long s = nb[row];
        if (s != curSeg) {
          if (curSeg >= 0) atomicMaxFloatBits(&segmax_bits[(int)curSeg], curMax);
          curSeg = s;
          curMax = p[v];
        } else {
          curMax = fmaxf(curMax, p[v]);
        }
      }
    }
    if (curSeg >= 0) atomicMaxFloatBits(&segmax_bits[(int)curSeg], curMax);
  }
}

// ---------------------------------------------------------------------------
// e = exp(logit - segmax); wave-aggregated segment sum
// ---------------------------------------------------------------------------
__global__ __launch_bounds__(256) void ca_exp_kernel(
    const long long* __restrict__ nb,
    float* __restrict__ buf,                 // in: logits, out: e (in place)
    const unsigned int* __restrict__ segmax_bits,
    float* __restrict__ segsum,
    int N)
{
  int i  = blockIdx.x * 256 + threadIdx.x;
  int ic = (i < N) ? i : (N - 1);
  int s  = (int)nb[ic];
  float lg = buf[ic];
  float m  = __uint_as_float(segmax_bits[s]);
  float e  = (i < N) ? __expf(lg - m) : 0.0f;   // arg <= 0, safe for v_exp_f32
  if (i < N) buf[ic] = e;

  int s0 = __shfl(s, 0, 32);
  bool same = __all(s == s0);
  if (same) {
    float sum = e;
    #pragma unroll
    for (int m2 = 16; m2 >= 1; m2 >>= 1) sum += __shfl_xor(sum, m2, 32);
    if ((threadIdx.x & 31) == 0) atomicAdd(&segsum[s0], sum);
  } else if (i < N) {
    atomicAdd(&segsum[s], e);
  }
}

// ---------------------------------------------------------------------------
// out = e / segsum[seg]
// ---------------------------------------------------------------------------
__global__ __launch_bounds__(256) void ca_norm_kernel(
    const long long* __restrict__ nb,
    const float* __restrict__ e,
    const float* __restrict__ segsum,
    float* __restrict__ out,
    int N)
{
  int i = blockIdx.x * 256 + threadIdx.x;
  if (i >= N) return;
  int s = (int)nb[i];
  out[i] = e[i] / segsum[s];
}

// ---------------------------------------------------------------------------
extern "C" void kernel_launch(void* const* d_in, const int* in_sizes, int n_in,
                              void* d_out, int out_size, void* d_ws, size_t ws_size,
                              hipStream_t stream)
{
  const float*     x    = (const float*)d_in[0];
  const long long* nb   = (const long long*)d_in[1];   // int64 per reference
  const float*     gfea = (const float*)d_in[2];
  const float*     W1   = (const float*)d_in[3];
  const float*     b1   = (const float*)d_in[4];
  const float*     W2   = (const float*)d_in[5];
  const float*     b2   = (const float*)d_in[6];
  float*           out  = (float*)d_out;

  const int N = in_sizes[0] / NEURONS;
  const int B = in_sizes[2] / GLOBAL_D;

  // workspace carve-up
  char* ws = (char*)d_ws;
  size_t off = 0;
  _Float16* gfeat = (_Float16*)(ws + off); off += (size_t)B * GFPAD * sizeof(_Float16);
  _Float16* w1s   = (_Float16*)(ws + off); off += (size_t)W1S_ELEMS * sizeof(_Float16);
  off = (off + 255) & ~(size_t)255;
  unsigned int* segmax = (unsigned int*)(ws + off); off += (size_t)B * sizeof(unsigned int);
  float*        segsum = (float*)(ws + off);        off += (size_t)B * sizeof(float);
  float*        logits = (float*)(ws + off);        off += (size_t)N * sizeof(float);

  const int prep_total = B + W1S_ELEMS + B * GFPAD;
  ca_prep_kernel<<<(prep_total + 255) / 256, 256, 0, stream>>>(
      W1, gfea, w1s, gfeat, segmax, segsum, B);

  const long long tiles = ((long long)N + 15) / 16;
  const int blocks1 = (int)((tiles + 7) / 8);
  ca_logits_kernel<<<blocks1, 256, 0, stream>>>(
      x, nb, gfeat, w1s, b1, W2, b2, logits, segmax, N);

  const int blocksN = (N + 255) / 256;
  ca_exp_kernel<<<blocksN, 256, 0, stream>>>(nb, logits, segmax, segsum, N);
  ca_norm_kernel<<<blocksN, 256, 0, stream>>>(nb, logits, segsum, out, N);
}
